// AttentionHead_33062658245001
// MI455X (gfx1250) — matrix operand order repaired and checked by reading, not compiled, not run on hardware
//
#include <hip/hip_runtime.h>
#include <hip/hip_bf16.h>

#define B_ 8
#define N_ 2048
#define D_ 512

typedef __bf16 bf16_t;
typedef __attribute__((ext_vector_type(16))) __bf16 v16bf;
typedef __attribute__((ext_vector_type(8)))  __bf16 v8bf;
typedef __attribute__((ext_vector_type(8)))  float  v8f;

#if __has_builtin(__builtin_amdgcn_global_load_async_to_lds_b128)
#define HAVE_ASYNC_LDS 1
typedef int vi4 __attribute__((ext_vector_type(4)));
typedef __attribute__((address_space(1))) vi4 gvi4;
typedef __attribute__((address_space(3))) vi4 lvi4;
#endif

// ---- fragment loaders (CDNA5 wave32 WMMA layouts) --------------------------
// A (16x32 bf16): lane l -> row M = l%16; K cols: l<16 -> {0..7,16..23},
//                 l>=16 -> {8..15,24..31}, packed 2 per VGPR.
static __device__ __forceinline__ v16bf load_a_frag(const bf16_t* __restrict__ row,
                                                    int kbase, int half) {
  v8bf c0 = *(const v8bf*)(row + kbase + half * 8);
  v8bf c1 = *(const v8bf*)(row + kbase + 16 + half * 8);
  v16bf f;
#pragma unroll
  for (int j = 0; j < 8; ++j) { f[j] = c0[j]; f[j + 8] = c1[j]; }
  return f;
}
// B (32x16 bf16): lane l -> col N = l%16; K rows: l<16 -> {0..7,16..23},
//                 l>=16 -> {8..15,24..31} (same k order as A).
static __device__ __forceinline__ v16bf load_b_frag(const bf16_t* __restrict__ row,
                                                    int kbase, int half) {
  v8bf c0 = *(const v8bf*)(row + kbase + half * 8);
  v8bf c1 = *(const v8bf*)(row + kbase + 16 + half * 8);
  v16bf f;
#pragma unroll
  for (int j = 0; j < 8; ++j) { f[j] = c0[j]; f[j + 8] = c1[j]; }
  return f;
}
static __device__ __forceinline__ v8f wmma_bf16(v16bf a, v16bf b, v8f c) {
  return __builtin_amdgcn_wmma_f32_16x16x32_bf16(false, a, false, b,
                                                 (short)0, c, false, false);
}

// Stage `n16` 16-byte chunks from global to LDS, whole-block cooperative.
static __device__ __forceinline__ void stage_to_lds(const bf16_t* __restrict__ gsrc,
                                                    bf16_t* __restrict__ ldst,
                                                    int n16) {
#ifdef HAVE_ASYNC_LDS
  gvi4* g = (gvi4*)(const void*)gsrc;
  lvi4* l = (lvi4*)(void*)ldst;
  for (int i = threadIdx.x; i < n16; i += (int)blockDim.x)
    __builtin_amdgcn_global_load_async_to_lds_b128(g + i, l + i, 0, 0);
#if __has_builtin(__builtin_amdgcn_s_wait_asynccnt)
  __builtin_amdgcn_s_wait_asynccnt(0);
#else
  asm volatile("s_wait_asynccnt 0x0" ::: "memory");
#endif
#else
  const uint4* s = (const uint4*)gsrc;
  uint4* d = (uint4*)ldst;
  for (int i = threadIdx.x; i < n16; i += (int)blockDim.x) d[i] = s[i];
#endif
}

// ---- fp32 -> bf16 conversion ----------------------------------------------
__global__ void cvt_f32_bf16(const float* __restrict__ src,
                             bf16_t* __restrict__ dst, int n) {
  int i = blockIdx.x * blockDim.x + threadIdx.x;
  if (i < n) dst[i] = (bf16_t)src[i];
}

// ---- QKV projection: Q,K row-major bf16; V stored transposed (D x N) -------
__global__ __launch_bounds__(128) void qkv_kernel(
    const bf16_t* __restrict__ Xbf,
    const bf16_t* __restrict__ Wqb, const bf16_t* __restrict__ Wkb,
    const bf16_t* __restrict__ Wvb,
    const float* __restrict__ bq, const float* __restrict__ bk,
    const float* __restrict__ bv,
    bf16_t* __restrict__ Qbf, bf16_t* __restrict__ Kbf, bf16_t* __restrict__ Vt) {
  const int lane = threadIdx.x & 31;
  const int wave = threadIdx.x >> 5;
  const int half = lane >> 4;
  const int l16  = lane & 15;
  const int b    = blockIdx.x >> 5;     // 32 strips of 64 rows per batch
  const int strip= blockIdx.x & 31;
  const int row0 = strip * 64 + wave * 16;

  const bf16_t* xrow = Xbf + ((size_t)b * N_ + row0 + l16) * D_;

  for (int which = 0; which < 3; ++which) {
    const bf16_t* W    = which == 0 ? Wqb : (which == 1 ? Wkb : Wvb);
    const float*  bias = which == 0 ? bq  : (which == 1 ? bk  : bv );
    const float   scl  = which == 0 ? 0.044194173824159216f : 1.0f; // 1/sqrt(512) folded into Q
    for (int et = 0; et < 32; ++et) {
      const bf16_t* wrow = W + (size_t)(et * 16 + l16) * D_;
      v8f acc;
#pragma unroll
      for (int r = 0; r < 8; ++r) acc[r] = 0.f;
#pragma unroll
      for (int dk = 0; dk < 16; ++dk) {
        v16bf a  = load_a_frag(xrow, dk * 32, half);   // A = X rows
        v16bf bb = load_b_frag(wrow, dk * 32, half);   // B = W^T (row load of W)
        acc = wmma_bf16(a, bb, acc);
      }
      const float bval = bias[et * 16 + l16];
      if (which < 2) {
        bf16_t* dst = (which == 0 ? Qbf : Kbf);
#pragma unroll
        for (int r = 0; r < 8; ++r) {
          const int m = r + 8 * half;   // C layout: M = r + 8*(lane/16), N = lane%16
          dst[((size_t)b * N_ + row0 + m) * D_ + et * 16 + l16] =
              (bf16_t)((acc[r] + bval) * scl);
        }
      } else {
#pragma unroll
        for (int r = 0; r < 8; ++r) {
          const int m = r + 8 * half;
          Vt[((size_t)b * D_ + et * 16 + l16) * N_ + row0 + m] =
              (bf16_t)(acc[r] + bval);
        }
      }
    }
  }
}

// ---- per-batch column mean of V (for fully-masked query rows) --------------
__global__ void vmean_kernel(const bf16_t* __restrict__ Vt,
                             float* __restrict__ vmean) {
  int idx = blockIdx.x * blockDim.x + threadIdx.x;
  if (idx >= B_ * D_) return;
  const bf16_t* row = Vt + (size_t)idx * N_;
  float s = 0.f;
  for (int n = 0; n < N_; n += 8) {
    v8bf v = *(const v8bf*)(row + n);
#pragma unroll
    for (int j = 0; j < 8; ++j) s += (float)v[j];
  }
  vmean[idx] = s * (1.0f / N_);
}

// ---- flash attention: S^T = K*Q^T so P lands directly in A-frag layout -----
__global__ __launch_bounds__(128) void attn_kernel(
    const bf16_t* __restrict__ Qbf, const bf16_t* __restrict__ Kbf,
    const bf16_t* __restrict__ Vt, const float* __restrict__ vmean,
    const long long* __restrict__ lens, float* __restrict__ out) {
  __shared__ bf16_t Qs[16 * D_];   // 16 KB
  __shared__ bf16_t Ks[32 * D_];   // 32 KB
  const int lane = threadIdx.x & 31;
  const int wv   = threadIdx.x >> 5;
  const int half = lane >> 4;
  const int l16  = lane & 15;
  const int b    = blockIdx.x >> 7;
  const int qt   = blockIdx.x & 127;
  const int row0 = qt * 16;
  const int cb   = wv * 128;                 // this wave's channel slice
  const int L    = (int)lens[b];

  // stage Q strip into LDS (direct-to-LDS async; visible after first barrier)
  stage_to_lds(Qbf + ((size_t)b * N_ + row0) * D_, Qs, 16 * D_ * 2 / 16);

  v8f O[8];
#pragma unroll
  for (int t = 0; t < 8; ++t)
#pragma unroll
    for (int r = 0; r < 8; ++r) O[t][r] = 0.f;
  float mstat = -1e30f, lstat = 0.f;

  for (int j0 = 0; j0 < L; j0 += 32) {
    __syncthreads();
    // cooperative direct-to-LDS stage of 32-key K chunk (32 KB)
    stage_to_lds(Kbf + ((size_t)b * N_ + j0) * D_, Ks, 32 * D_ * 2 / 16);
    __syncthreads();

    if (j0 + 32 < L) { // prefetch next K chunk (global_prefetch_b8)
      const char* nk = (const char*)(Kbf + ((size_t)b * N_ + j0 + 32) * D_);
      __builtin_prefetch(nk + (size_t)threadIdx.x * 256, 0, 1);
      __builtin_prefetch(nk + (size_t)threadIdx.x * 256 + 128, 0, 1);
    }

    // S^T tiles: M = key (VGPRs), N = query (lanes)
    v8f s0, s1;
#pragma unroll
    for (int r = 0; r < 8; ++r) { s0[r] = 0.f; s1[r] = 0.f; }
    const bf16_t* qrow  = Qs + l16 * D_;
    const bf16_t* k0row = Ks + l16 * D_;
    const bf16_t* k1row = Ks + (16 + l16) * D_;
#pragma unroll
    for (int dk = 0; dk < 16; ++dk) {
      v16bf bqf = load_b_frag(qrow, dk * 32, half);   // B = Q^T (row load of Q)
      v16bf a0  = load_a_frag(k0row, dk * 32, half);  // A = K rows 0..15
      s0 = wmma_bf16(a0, bqf, s0);
      v16bf a1  = load_a_frag(k1row, dk * 32, half);  // A = K rows 16..31
      s1 = wmma_bf16(a1, bqf, s1);
    }

    // online softmax; query = l16 (replicated in lanes l and l^16)
    float p0[8], p1[8];
    float cmax = -1e30f;
#pragma unroll
    for (int r = 0; r < 8; ++r) {
      p0[r] = (j0 + r + 8 * half      < L) ? s0[r] : -1e30f;
      p1[r] = (j0 + 16 + r + 8 * half < L) ? s1[r] : -1e30f;
      cmax = fmaxf(cmax, fmaxf(p0[r], p1[r]));
    }
    cmax = fmaxf(cmax, __shfl_xor(cmax, 16, 32));
    const float mnew  = fmaxf(mstat, cmax);
    const float scale = __expf(mstat - mnew);
    float rsum = 0.f;
#pragma unroll
    for (int r = 0; r < 8; ++r) {
      p0[r] = __expf(p0[r] - mnew);
      p1[r] = __expf(p1[r] - mnew);
      rsum += p0[r] + p1[r];
    }
    rsum += __shfl_xor(rsum, 16, 32);
    lstat = lstat * scale + rsum;
    mstat = mnew;

    // broadcast per-query rescale into O's row (VGPR) dimension
    float osc[8];
#pragma unroll
    for (int r = 0; r < 8; ++r) osc[r] = __shfl(scale, half * 8 + r, 32);

    // pack P into A-fragment (free transpose thanks to S^T)
    v16bf pf;
#pragma unroll
    for (int j = 0; j < 8; ++j) { pf[j] = (bf16_t)p0[j]; pf[j + 8] = (bf16_t)p1[j]; }

#pragma unroll
    for (int t = 0; t < 8; ++t) {
      const bf16_t* vrow = Vt + ((size_t)b * D_ + cb + t * 16 + l16) * N_;
      v16bf bvf = load_b_frag(vrow, j0, half);   // B = V (contiguous keys in Vt row)
      v8f o = O[t];
#pragma unroll
      for (int r = 0; r < 8; ++r) o[r] *= osc[r];
      o = wmma_bf16(pf, bvf, o);
      O[t] = o;
    }
  }

  const float linv = 1.0f / lstat;
  float li[8];
#pragma unroll
  for (int r = 0; r < 8; ++r) li[r] = __shfl(linv, half * 8 + r, 32);

#pragma unroll
  for (int t = 0; t < 8; ++t) {
    const int c = cb + t * 16 + l16;
#pragma unroll
    for (int r = 0; r < 8; ++r) {
      const int q = row0 + r + 8 * half;
      float val = (q < L) ? O[t][r] * li[r] : vmean[b * D_ + c];
      out[((size_t)b * N_ + q) * D_ + c] = val;
    }
  }
}

extern "C" void kernel_launch(void* const* d_in, const int* in_sizes, int n_in,
                              void* d_out, int out_size, void* d_ws, size_t ws_size,
                              hipStream_t stream) {
  (void)in_sizes; (void)n_in; (void)out_size; (void)ws_size;
  const float*     x    = (const float*)d_in[0];
  const long long* lens = (const long long*)d_in[1];
  const float*     Wq   = (const float*)d_in[2];
  const float*     bq   = (const float*)d_in[3];
  const float*     Wk   = (const float*)d_in[4];
  const float*     bk   = (const float*)d_in[5];
  const float*     Wv   = (const float*)d_in[6];
  const float*     bv   = (const float*)d_in[7];
  float* out = (float*)d_out;

  char* ws = (char*)d_ws;
  const size_t xN = (size_t)B_ * N_ * D_;
  const size_t wN = (size_t)D_ * D_;
  bf16_t* Xbf = (bf16_t*)ws; ws += xN * 2;
  bf16_t* Qbf = (bf16_t*)ws; ws += xN * 2;
  bf16_t* Kbf = (bf16_t*)ws; ws += xN * 2;
  bf16_t* Vt  = (bf16_t*)ws; ws += xN * 2;
  bf16_t* Wqb = (bf16_t*)ws; ws += wN * 2;
  bf16_t* Wkb = (bf16_t*)ws; ws += wN * 2;
  bf16_t* Wvb = (bf16_t*)ws; ws += wN * 2;
  float*  vmean = (float*)ws;             // + B_*D_*4

  cvt_f32_bf16<<<(int)((xN + 255) / 256), 256, 0, stream>>>(x, Xbf, (int)xN);
  cvt_f32_bf16<<<(int)((wN + 255) / 256), 256, 0, stream>>>(Wq, Wqb, (int)wN);
  cvt_f32_bf16<<<(int)((wN + 255) / 256), 256, 0, stream>>>(Wk, Wkb, (int)wN);
  cvt_f32_bf16<<<(int)((wN + 255) / 256), 256, 0, stream>>>(Wv, Wvb, (int)wN);

  qkv_kernel<<<B_ * (N_ / 64), 128, 0, stream>>>(Xbf, Wqb, Wkb, Wvb,
                                                 bq, bk, bv, Qbf, Kbf, Vt);
  vmean_kernel<<<(B_ * D_ + 255) / 256, 256, 0, stream>>>(Vt, vmean);
  attn_kernel<<<B_ * (N_ / 16), 128, 0, stream>>>(Qbf, Kbf, Vt, vmean, lens, out);
}
